// STFT_29789893165288
// MI455X (gfx1250) — matrix operand order repaired and checked by reading, not compiled, not run on hardware
//
#include <hip/hip_runtime.h>

// ---------------- problem constants ----------------
#define NFFT      2048
#define HOP       512
#define ROWS      1025          // 1 + NFFT/2
#define BATCH     16
#define LEN       320000
#define TFRAMES   626
#define MTOT      (BATCH*TFRAMES)      // 10016
#define XP        (LEN + NFFT)         // 322048 padded length per batch
#define OUT_PLANE ((long)MTOT*ROWS)    // 10,266,400 floats (real plane)

#define MSUB      3                    // M-subtiles per wave (48 rows)
#define BLK_M     (8*16*MSUB)          // 384 rows per block (8 waves)

typedef __bf16 bf16;
typedef __attribute__((ext_vector_type(8)))  bf16  bf16x8;
typedef __attribute__((ext_vector_type(16))) bf16  bf16x16;
typedef __attribute__((ext_vector_type(8)))  float f32x8;

// ---------------- helpers ----------------
__device__ __forceinline__ unsigned short f2bf_rn(float f) {
    unsigned int u = __float_as_uint(f);
    u += 0x7FFFu + ((u >> 16) & 1u);          // round-to-nearest-even
    return (unsigned short)(u >> 16);
}
__device__ __forceinline__ float bf2f(unsigned short h) {
    return __uint_as_float(((unsigned int)h) << 16);
}

// Reflect-padded x -> interleaved hi/lo bf16 plane:
// element j of padded signal lives at xc[(j>>3)*16 + (j&7)] (hi) and +8 (lo).
__global__ __launch_bounds__(256) void prep_pad(const float* __restrict__ x,
                                                unsigned short* __restrict__ xc) {
    int idx = blockIdx.x * blockDim.x + threadIdx.x;
    if (idx >= BATCH * XP) return;
    int b = idx / XP;
    int j = idx - b * XP;
    int i = j - (NFFT / 2);
    i = (i < 0) ? -i : i;                      // left reflect
    if (i >= LEN) i = 2 * (LEN - 1) - i;       // right reflect
    float v = x[b * LEN + i];
    unsigned short h = f2bf_rn(v);
    unsigned c = ((unsigned)idx >> 3) * 16u + ((unsigned)idx & 7u);
    xc[c]     = h;
    xc[c + 8] = f2bf_rn(v - bf2f(h));
}

// Weights -> combined interleaved layout:
// wc[(r*64 + (k>>5))*128 + plane*32 + (k&31)], planes: 0=re-hi,1=re-lo,2=im-hi,3=im-lo.
__global__ __launch_bounds__(256) void prep_w(const float* __restrict__ w,
                                              unsigned short* __restrict__ wc,
                                              int p0) {
    int idx = blockIdx.x * blockDim.x + threadIdx.x;
    if (idx >= ROWS * NFFT) return;
    int r = idx >> 11;
    int k = idx & (NFFT - 1);
    float v = w[idx];
    unsigned short h = f2bf_rn(v);
    unsigned c = ((unsigned)(r * 64 + (k >> 5))) * 128u + (unsigned)(k & 31);
    wc[c + (unsigned)p0 * 32u]       = h;
    wc[c + (unsigned)(p0 + 1) * 32u] = f2bf_rn(v - bf2f(h));
}

#define WMMA_BF16(A, B, C) \
    __builtin_amdgcn_wmma_f32_16x16x32_bf16(false, (A), false, (B), (short)0, (C), false, false)

#define SHUF16(a, b) __builtin_shufflevector((a), (b), 0,1,2,3,4,5,6,7,8,9,10,11,12,13,14,15)

// ---------------- main GEMM kernel ----------------
// Block: 256 threads = 8 waves. Block tile: M=384 (48 rows/wave), N=32.
// Each wave: 3 M-subtiles x 2 N-subtiles x {real,imag} = 12 f32x8 accumulators.
// bf16x3: acc += Ah*Bh + Ah*Bl + Al*Bh per K-step of 32.
// One load-base per M-subtile (4 loads, imm {0,16,32,48}) and per N-subtile
// (8 loads, imm {0,16,...,208}) -> only 5 live address pairs in the K-loop.
__global__ __launch_bounds__(256) void stft_wmma(
    const bf16* __restrict__ xc, const bf16* __restrict__ wc,
    float* __restrict__ out)
{
    const int lane = threadIdx.x & 31;
    const int wave = threadIdx.x >> 5;
    const int hi16 = (lane >> 4) & 1;     // 0: lanes 0-15, 1: lanes 16-31
    const int l15  = lane & 15;

    const int n0    = blockIdx.x * 32;               // frequency-tile base
    const int mwave = blockIdx.y * BLK_M + wave * (16 * MSUB);

    // A lane offsets (elements in xc, loop-invariant) per M-subtile.
    unsigned av[MSUB];
#pragma unroll
    for (int ms = 0; ms < MSUB; ++ms) {
        int m  = mwave + ms * 16 + l15;
        int mc = (m < MTOT - 1) ? m : (MTOT - 1);   // clamp tail rows (stores predicated)
        int b  = mc / TFRAMES;
        int t  = mc - b * TFRAMES;
        av[ms] = 2u * (unsigned)(b * XP + t * HOP + hi16 * 16);
    }
    // B lane offsets (elements in wc, loop-invariant) per N-subtile.
    unsigned bv[2];
#pragma unroll
    for (int ns = 0; ns < 2; ++ns) {
        int r  = n0 + ns * 16 + l15;
        int rc = (r < ROWS - 1) ? r : (ROWS - 1);
        bv[ns] = (unsigned)rc * 8192u + (unsigned)hi16 * 16u;
    }

    f32x8 acc[MSUB][2][2] = {};   // [msub][nsub][real/imag]

    for (int k0 = 0; k0 < NFFT; k0 += 32) {
        const bf16* pA = xc + 2 * k0;   // uniform bases for this K-step
        const bf16* pB = wc + 4 * k0;

        // A fragments: hi/lo interleaved, one base per subtile.
        bf16x16 Ah[MSUB], Al[MSUB];
#pragma unroll
        for (int ms = 0; ms < MSUB; ++ms) {
            const bf16* p = pA + av[ms];
            bf16x8 h0 = *(const bf16x8*)(p);
            bf16x8 l0 = *(const bf16x8*)(p + 8);
            bf16x8 h1 = *(const bf16x8*)(p + 16);
            bf16x8 l1 = *(const bf16x8*)(p + 24);
            Ah[ms] = SHUF16(h0, h1);
            Al[ms] = SHUF16(l0, l1);
        }

#pragma unroll
        for (int ns = 0; ns < 2; ++ns) {
            const bf16* p = pB + bv[ns];
            bf16x8 rh0 = *(const bf16x8*)(p);          // plane 0: re-hi
            bf16x8 rh1 = *(const bf16x8*)(p + 8);
            bf16x8 rl0 = *(const bf16x8*)(p + 32);     // plane 1: re-lo
            bf16x8 rl1 = *(const bf16x8*)(p + 40);
            bf16x8 ih0 = *(const bf16x8*)(p + 64);     // plane 2: im-hi
            bf16x8 ih1 = *(const bf16x8*)(p + 72);
            bf16x8 il0 = *(const bf16x8*)(p + 96);     // plane 3: im-lo
            bf16x8 il1 = *(const bf16x8*)(p + 104);
            bf16x16 Brh = SHUF16(rh0, rh1);
            bf16x16 Brl = SHUF16(rl0, rl1);
            bf16x16 Bih = SHUF16(ih0, ih1);
            bf16x16 Bil = SHUF16(il0, il1);

            // term 1: Ah*Bh  (6 independent WMMAs before any acc reuse)
#pragma unroll
            for (int ms = 0; ms < MSUB; ++ms) {
                acc[ms][ns][0] = WMMA_BF16(Ah[ms], Brh, acc[ms][ns][0]);
                acc[ms][ns][1] = WMMA_BF16(Ah[ms], Bih, acc[ms][ns][1]);
            }
            // term 2: Ah*Bl
#pragma unroll
            for (int ms = 0; ms < MSUB; ++ms) {
                acc[ms][ns][0] = WMMA_BF16(Ah[ms], Brl, acc[ms][ns][0]);
                acc[ms][ns][1] = WMMA_BF16(Ah[ms], Bil, acc[ms][ns][1]);
            }
            // term 3: Al*Bh
#pragma unroll
            for (int ms = 0; ms < MSUB; ++ms) {
                acc[ms][ns][0] = WMMA_BF16(Al[ms], Brh, acc[ms][ns][0]);
                acc[ms][ns][1] = WMMA_BF16(Al[ms], Bih, acc[ms][ns][1]);
            }
        }
    }

    // Store: C/D layout -> element v is row (v + hi16*8), col (lane&15).
#pragma unroll
    for (int ms = 0; ms < MSUB; ++ms) {
        int mbase = mwave + ms * 16 + hi16 * 8;
#pragma unroll
        for (int ns = 0; ns < 2; ++ns) {
            int r = n0 + ns * 16 + l15;
            if (r < ROWS) {
#pragma unroll
                for (int v = 0; v < 8; ++v) {
                    int m = mbase + v;
                    if (m < MTOT) {
                        out[(long)m * ROWS + r]             = acc[ms][ns][0][v];
                        out[OUT_PLANE + (long)m * ROWS + r] = acc[ms][ns][1][v];
                    }
                }
            }
        }
    }
}

// ---------------- launcher ----------------
extern "C" void kernel_launch(void* const* d_in, const int* in_sizes, int n_in,
                              void* d_out, int out_size, void* d_ws, size_t ws_size,
                              hipStream_t stream) {
    const float* x  = (const float*)d_in[0];
    const float* wr = (const float*)d_in[1];
    const float* wi = (const float*)d_in[2];

    char* ws = (char*)d_ws;
    const size_t XCB = (size_t)BATCH * XP * 2 * 2;  // interleaved hi/lo x plane bytes
    unsigned short* xc = (unsigned short*)(ws);
    unsigned short* wcm = (unsigned short*)(ws + XCB);  // 4-plane interleaved weights

    const int np = BATCH * XP;
    prep_pad<<<(np + 255) / 256, 256, 0, stream>>>(x, xc);
    const int nw = ROWS * NFFT;
    prep_w<<<(nw + 255) / 256, 256, 0, stream>>>(wr, wcm, 0);
    prep_w<<<(nw + 255) / 256, 256, 0, stream>>>(wi, wcm, 2);

    dim3 grid((ROWS + 31) / 32, (MTOT + BLK_M - 1) / BLK_M);   // 33 x 27 blocks
    stft_wmma<<<grid, dim3(256), 0, stream>>>(
        (const bf16*)xc, (const bf16*)wcm, (float*)d_out);
}
